// RecurrentGemmaRglru_8134668058990
// MI455X (gfx1250) — matrix operand order repaired and checked by reading, compiled, not run on hardware
//
#include <hip/hip_runtime.h>
#include <hip/hip_bf16.h>
#include <math.h>

// ---------------------------------------------------------------------------
// RG-LRU (RecurrentGemma) for MI455X / gfx1250.
//   Phase 0: one-time pre-pack of gate weights into WMMA B-fragment layout,
//            split-precision bf16 (hi + lo), ~5 MB (L2-resident).
//   Phase 1: per-head gate GEMMs via V_WMMA_F32_16X16X32_BF16; A tile staged
//            through LDS directly in A-fragment layout (b128 reads), B
//            fragments are straight b128 loads of the packed weights.
//            Fused sigmoid/softplus/exp epilogue produces a[], xnorm[].
//   Phase 2: chunked parallel linear scan  h = a*h + x  (local scan,
//            carry combine, fixup) for high occupancy (640 blocks).
// ---------------------------------------------------------------------------

#define B_N      4
#define S_LEN    2048
#define D_DIM    2560
#define H_N      10
#define BW       256
#define T_TOK    (B_N * S_LEN)          // 8192 tokens
#define MTILES   (T_TOK / 16)           // 512
#define NCHUNK   16
#define CHUNK    (S_LEN / NCHUNK)       // 128
#define BD       (B_N * D_DIM)          // 10240
#define NTILES   (BW / 16)              // 16 N-tiles per head
#define KTILES   (BW / 32)              // 8 K-steps per GEMM
#define PACK_SLOTS (H_N * NTILES * KTILES * 32)   // 40960 lane-slots
#define PACK_ELEMS ((size_t)PACK_SLOTS * 16)      // 655360 bf16 per array

typedef __attribute__((ext_vector_type(16))) __bf16 v16bf;
typedef __attribute__((ext_vector_type(8)))  float  v8f;

// ---------------------------------------------------------------------------
// Phase 0: pack both gate weight matrices into per-lane WMMA B fragments.
// B-fragment layout (contiguous k order used here, same as A): lane l owns column n = nBase + (l&15);
// element e of the v16bf holds K = kt*32 + ((l>=16)?16:0) + e.
// Output: packed[((h*NTILES + nt)*KTILES + kt)*32 + l][e], hi & lo split.
// ---------------------------------------------------------------------------
__global__ __launch_bounds__(256) void rglru_pack_weights(
    const float* __restrict__ wi, const float* __restrict__ wr,
    __bf16* __restrict__ wiHi, __bf16* __restrict__ wiLo,
    __bf16* __restrict__ wrHi, __bf16* __restrict__ wrLo)
{
    const int t = blockIdx.x * 256 + threadIdx.x;   // one lane-slot per thread
    if (t >= PACK_SLOTS) return;
    const int l   = t & 31;
    const int kt  = (t >> 5) & 7;
    const int nt  = (t >> 8) & 15;
    const int h   = t >> 12;

    const int n     = nt * 16 + (l & 15);
    const int kHalf = (l >> 4) * 16;
    const float* wiCol = wi + (size_t)h * BW * BW + n;
    const float* wrCol = wr + (size_t)h * BW * BW + n;

    v16bf ihi, ilo, rhi, rlo;
    #pragma unroll
    for (int e = 0; e < 16; ++e) {
        const int k = kt * 32 + kHalf + e;
        const float vi = wiCol[(size_t)k * BW];
        const float vr = wrCol[(size_t)k * BW];
        __bf16 th = (__bf16)vi;
        ihi[e] = th; ilo[e] = (__bf16)(vi - (float)th);
        th = (__bf16)vr;
        rhi[e] = th; rlo[e] = (__bf16)(vr - (float)th);
    }
    const size_t o = (size_t)t * 16;
    *reinterpret_cast<v16bf*>(wiHi + o) = ihi;
    *reinterpret_cast<v16bf*>(wiLo + o) = ilo;
    *reinterpret_cast<v16bf*>(wrHi + o) = rhi;
    *reinterpret_cast<v16bf*>(wrLo + o) = rlo;
}

// ---------------------------------------------------------------------------
// Phase 1: WMMA gate kernel.
// grid = (MTILES, H_N), block = 256 threads (8 waves of 32).
// LDS holds the 16x256 activation tile pre-split into bf16 hi/lo, stored in
// A-fragment layout [kt][lane][16] so each fragment is two ds_load_b128.
// ---------------------------------------------------------------------------
__global__ __launch_bounds__(256) void rglru_gates_wmma(
    const float* __restrict__ act,      // [T_TOK, D_DIM]
    const int*   __restrict__ pos,      // [T_TOK]
    const float* __restrict__ rparam,   // [D_DIM]
    const float* __restrict__ bi,       // [H_N, BW]
    const float* __restrict__ br,       // [H_N, BW]
    const __bf16* __restrict__ wiHi, const __bf16* __restrict__ wiLo,
    const __bf16* __restrict__ wrHi, const __bf16* __restrict__ wrLo,
    float* __restrict__ aOut,           // [T_TOK, D_DIM]
    float* __restrict__ xOut)           // [T_TOK, D_DIM]
{
    const int mtile   = blockIdx.x;
    const int h       = blockIdx.y;
    const int tid     = threadIdx.x;
    const int tokBase = mtile * 16;

    // A fragments: [kt][lane][e], 32 bytes per lane-slot, conflict-free b128.
    __shared__ __align__(32) __bf16 AhiF[KTILES][32][16];
    __shared__ __align__(32) __bf16 AloF[KTILES][32][16];

    // Cooperative load + split + scatter into fragment layout.
    // Thread tid covers column K=tid of rows 0..15 (coalesced global reads).
    {
        const int ktc  = tid >> 5;       // K-tile of this column
        const int kRem = tid & 31;
        const int sub  = kRem & 7;
        const int grp  = kRem >> 3;      // 0..3
        const int laneAdd = (grp >= 2) ? 16 : 0;
        const int eOff    = (grp & 1) ? 8 : 0;
        #pragma unroll
        for (int m = 0; m < 16; ++m) {
            const float v = act[(size_t)(tokBase + m) * D_DIM + h * BW + tid];
            const __bf16 hi = (__bf16)v;
            AhiF[ktc][m + laneAdd][eOff + sub] = hi;
            AloF[ktc][m + laneAdd][eOff + sub] = (__bf16)(v - (float)hi);
        }
    }
    __syncthreads();

    const int wave   = tid >> 5;
    const int lane   = tid & 31;
    const int laneM  = lane & 15;        // B,C column within tile
    const int hiHalf = lane >> 4;        // 0: lanes 0-15, 1: lanes 16-31

    for (int nt = 0; nt < 2; ++nt) {
        const int ntile = wave * 2 + nt;
        const int nBase = ntile * 16;
        const int n     = nBase + laneM;                 // output channel in head
        // Packed-fragment base for this (h, ntile): advance by 32 slots per kt.
        const size_t fBase = ((size_t)(h * NTILES + ntile) * KTILES * 32 + lane) * 16;

        v8f accI = {};
        v8f accR = {};

        #pragma unroll
        for (int kt = 0; kt < KTILES; ++kt) {
            const size_t fo = fBase + (size_t)kt * 32 * 16;

            if (kt < KTILES - 1) {   // -> global_prefetch_b8 for next K-step
                __builtin_prefetch(wiHi + fo + 32 * 16, 0, 1);
                __builtin_prefetch(wrHi + fo + 32 * 16, 0, 1);
            }

            const v16bf aHi  = *reinterpret_cast<const v16bf*>(&AhiF[kt][lane][0]);
            const v16bf aLo  = *reinterpret_cast<const v16bf*>(&AloF[kt][lane][0]);
            const v16bf biH  = *reinterpret_cast<const v16bf*>(wiHi + fo);
            const v16bf biL  = *reinterpret_cast<const v16bf*>(wiLo + fo);
            const v16bf brH  = *reinterpret_cast<const v16bf*>(wrHi + fo);
            const v16bf brL  = *reinterpret_cast<const v16bf*>(wrLo + fo);

            // Split-precision: acc += Ahi*Bhi + Ahi*Blo + Alo*Bhi
            accI = __builtin_amdgcn_wmma_f32_16x16x32_bf16(false, aHi, false, biH, (short)0, accI, false, false);
            accI = __builtin_amdgcn_wmma_f32_16x16x32_bf16(false, aHi, false, biL, (short)0, accI, false, false);
            accI = __builtin_amdgcn_wmma_f32_16x16x32_bf16(false, aLo, false, biH, (short)0, accI, false, false);
            accR = __builtin_amdgcn_wmma_f32_16x16x32_bf16(false, aHi, false, brH, (short)0, accR, false, false);
            accR = __builtin_amdgcn_wmma_f32_16x16x32_bf16(false, aHi, false, brL, (short)0, accR, false, false);
            accR = __builtin_amdgcn_wmma_f32_16x16x32_bf16(false, aLo, false, brH, (short)0, accR, false, false);
        }

        // Epilogue: bias + sigmoid + rg-lru coefficients.
        const float biasI = bi[h * BW + n];
        const float biasR = br[h * BW + n];
        const float rp    = rparam[h * BW + n];
        const float sp    = fmaxf(rp, 0.f) + log1pf(expf(-fabsf(rp)));  // softplus

        #pragma unroll
        for (int r = 0; r < 8; ++r) {
            const int tok    = tokBase + hiHalf * 8 + r;    // C layout: M = r + 8*hiHalf
            const size_t idx = (size_t)tok * D_DIM + h * BW + n;

            const float ig    = 1.f / (1.f + expf(-(accI[r] + biasI)));
            const float rg    = 1.f / (1.f + expf(-(accR[r] + biasR)));
            const float log_a = -8.f * rg * sp;
            const float a     = expf(log_a);
            const float a2    = expf(2.f * log_a);
            const bool reset  = (pos[tok] == 0);

            const float xval = act[idx];
            const float mult = reset ? 1.f : sqrtf(fmaxf(1.f - a2, 0.f));
            xOut[idx] = xval * ig * mult;
            aOut[idx] = reset ? 0.f : a;
        }
    }
}

// ---------------------------------------------------------------------------
// Phase 2a: chunk-local scan. One thread per channel d, one block per
// (b, d-block-of-256, chunk). Writes local y into d_out; records chunk-end
// state (yLast) and running product (pLast).
// ---------------------------------------------------------------------------
__global__ __launch_bounds__(256) void rglru_scan_local(
    const float* __restrict__ aA, const float* __restrict__ xA,
    float* __restrict__ out, float* __restrict__ yLast, float* __restrict__ pLast)
{
    const int blk  = blockIdx.x;
    const int c    = blk % NCHUNK;
    const int rem  = blk / NCHUNK;
    const int dblk = rem % (D_DIM / 256);
    const int b    = rem / (D_DIM / 256);
    const int d    = dblk * 256 + threadIdx.x;

    size_t base = ((size_t)b * S_LEN + (size_t)c * CHUNK) * D_DIM + d;
    float hloc = 0.f, P = 1.f;
    for (int s = 0; s < CHUNK; ++s) {
        const float av = aA[base];
        const float xv = xA[base];
        hloc = fmaf(av, hloc, xv);
        P *= av;
        out[base] = hloc;
        base += D_DIM;
    }
    const size_t bd = (size_t)b * D_DIM + d;
    yLast[bd * NCHUNK + c] = hloc;
    pLast[bd * NCHUNK + c] = P;
}

// ---------------------------------------------------------------------------
// Phase 2b: sequential carry combine across chunks (tiny: BD threads).
// ---------------------------------------------------------------------------
__global__ __launch_bounds__(256) void rglru_scan_carry(
    const float* __restrict__ yLast, const float* __restrict__ pLast,
    float* __restrict__ carryIn)
{
    const int bd = blockIdx.x * blockDim.x + threadIdx.x;
    if (bd >= BD) return;
    float carry = 0.f;
    #pragma unroll
    for (int c = 0; c < NCHUNK; ++c) {
        carryIn[(size_t)bd * NCHUNK + c] = carry;
        carry = yLast[(size_t)bd * NCHUNK + c] + pLast[(size_t)bd * NCHUNK + c] * carry;
    }
}

// ---------------------------------------------------------------------------
// Phase 2c: fixup  y_t += P_t * carry.
// ---------------------------------------------------------------------------
__global__ __launch_bounds__(256) void rglru_scan_fixup(
    const float* __restrict__ aA, const float* __restrict__ carryIn,
    float* __restrict__ out)
{
    const int blk  = blockIdx.x;
    const int c    = blk % NCHUNK;
    if (c == 0) return;                     // carry into chunk 0 is always 0
    const int rem  = blk / NCHUNK;
    const int dblk = rem % (D_DIM / 256);
    const int b    = rem / (D_DIM / 256);
    const int d    = dblk * 256 + threadIdx.x;

    const size_t bd = (size_t)b * D_DIM + d;
    const float carry = carryIn[bd * NCHUNK + c];
    if (carry == 0.f) return;               // data-dependent only: deterministic

    size_t base = ((size_t)b * S_LEN + (size_t)c * CHUNK) * D_DIM + d;
    float P = 1.f;
    for (int s = 0; s < CHUNK; ++s) {
        P *= aA[base];
        out[base] = fmaf(P, carry, out[base]);
        base += D_DIM;
    }
}

// ---------------------------------------------------------------------------
// Launch. Workspace layout (floats unless noted):
//   [0, ND)          a[]          (ND = T_TOK*D_DIM)
//   [ND, 2*ND)       xnorm[]
//   yLast, pLast, carryIn         (BD*NCHUNK each)
//   packed weights: wiHi, wiLo, wrHi, wrLo  (PACK_ELEMS bf16 each, 32B-aligned)
// Total ~169 MB + 5.3 MB.
// ---------------------------------------------------------------------------
extern "C" void kernel_launch(void* const* d_in, const int* in_sizes, int n_in,
                              void* d_out, int out_size, void* d_ws, size_t ws_size,
                              hipStream_t stream) {
    const float* act    = (const float*)d_in[0];
    const int*   pos    = (const int*)d_in[1];
    const float* rparam = (const float*)d_in[2];
    const float* wi     = (const float*)d_in[3];
    const float* bi     = (const float*)d_in[4];
    const float* wr     = (const float*)d_in[5];
    const float* br     = (const float*)d_in[6];
    float*       out    = (float*)d_out;

    const size_t ND = (size_t)T_TOK * D_DIM;
    float* wsA     = (float*)d_ws;
    float* wsX     = wsA + ND;
    float* yLast   = wsX + ND;
    float* pLast   = yLast + (size_t)BD * NCHUNK;
    float* carryIn = pLast + (size_t)BD * NCHUNK;
    __bf16* wiHi   = (__bf16*)(carryIn + (size_t)BD * NCHUNK);
    __bf16* wiLo   = wiHi + PACK_ELEMS;
    __bf16* wrHi   = wiLo + PACK_ELEMS;
    __bf16* wrLo   = wrHi + PACK_ELEMS;

    // Phase 0: pack weights into WMMA fragment layout (one-time, ~5 MB).
    rglru_pack_weights<<<(PACK_SLOTS + 255) / 256, 256, 0, stream>>>(
        wi, wr, wiHi, wiLo, wrHi, wrLo);

    // Phase 1: gates (WMMA).
    dim3 gGates(MTILES, H_N);
    rglru_gates_wmma<<<gGates, 256, 0, stream>>>(
        act, pos, rparam, bi, br, wiHi, wiLo, wrHi, wrLo, wsA, wsX);

    // Phase 2: chunked scan.
    const int scanBlocks = B_N * (D_DIM / 256) * NCHUNK;   // 640
    rglru_scan_local<<<scanBlocks, 256, 0, stream>>>(wsA, wsX, out, yLast, pLast);
    rglru_scan_carry<<<(BD + 255) / 256, 256, 0, stream>>>(yLast, pLast, carryIn);
    rglru_scan_fixup<<<scanBlocks, 256, 0, stream>>>(wsA, carryIn, out);
}
